// FastOriginalModel_70205535420774
// MI455X (gfx1250) — matrix operand order repaired and checked by reading, not compiled, not run on hardware
//
#include <hip/hip_runtime.h>
#include <math.h>

// ---------------------------------------------------------------------------
// Model dims (from reference)
// ---------------------------------------------------------------------------
#define B_     16
#define L_     512
#define HID_   768
#define NH_    12
#define DH_    64
#define FFN_   3072
#define NL_    6
#define CHUNK_ 256
#define BL_    (B_ * L_)      // 8192 tokens
#define ALPHA_ 0.6f

typedef _Float16 v16h __attribute__((ext_vector_type(16)));
typedef float    v8f  __attribute__((ext_vector_type(8)));

__device__ __forceinline__ float fast_sigmoid(float x) {
  return __builtin_amdgcn_rcpf(1.0f + __expf(-x));
}

// ---------------------------------------------------------------------------
// Block-wide sum reduction (256 threads). Ends with a barrier so `red` is
// immediately reusable.
// ---------------------------------------------------------------------------
__device__ __forceinline__ float block_reduce_sum(float v, float* red) {
  const int tid = threadIdx.x;
  red[tid] = v;
  __syncthreads();
  for (int s = 128; s > 0; s >>= 1) {
    if (tid < s) red[tid] += red[tid + s];
    __syncthreads();
  }
  float r = red[0];
  __syncthreads();
  return r;
}

// ---------------------------------------------------------------------------
// WMMA GEMM: C[M,N] = act(A[M,K] @ W[K,N] + bias)
// fp32 global -> f16 LDS -> v_wmma_f32_16x16x32_f16 -> fp32 out.
// Block tile 128(M) x 128(N), BK=32, double-buffered LDS. 256 threads =
// 8 waves arranged 4(M)x2(N); each wave owns 32x64 = 2x4 WMMA per K-step.
// A stored [m][k] (fragment = one contiguous 16-half run -> 2x ds_load_b128).
// W stored TRANSPOSED [n][k] so a B fragment (K = koff..koff+15 at fixed n)
// is one contiguous 16-half run -> 2x ds_load_b128 (no scalar re-pack).
// Pipeline: issue next tile's 8 global_load_b128 into distinct regs at the
// top of the iteration, run the 8 WMMAs, then cvt+store into the alternate
// LDS buffer; one barrier per iteration.
// M%128==0, N%128==0, K%32==0 always hold (M=8192, N/K in {768,3072}).
// ACT: 0 = none, 1 = silu, 2 = exact gelu.
// ---------------------------------------------------------------------------
#define BM 128
#define BN 128
#define BK 32

__device__ __forceinline__ void gemm_load_tile_regs(
    const float* __restrict__ A, const float* __restrict__ W,
    int rowBase, int colBase, int K, int N, int kb, int tid,
    float4 ra[4], float4 rw[4]) {
#pragma unroll
  for (int p = 0; p < 4; ++p) {
    const int m  = p * 32 + (tid >> 3);
    const int kc = (tid & 7) * 4;
    ra[p] = *reinterpret_cast<const float4*>(A + (size_t)(rowBase + m) * K + kb + kc);
  }
#pragma unroll
  for (int p = 0; p < 4; ++p) {
    const int r = p * 8 + (tid >> 5);
    const int c = (tid & 31) * 4;
    rw[p] = *reinterpret_cast<const float4*>(W + (size_t)(kb + r) * N + colBase + c);
  }
}

__device__ __forceinline__ void gemm_store_tile_lds(
    _Float16 (*__restrict__ As)[BK + 8], _Float16 (*__restrict__ WsT)[BK + 8],
    int tid, const float4 ra[4], const float4 rw[4]) {
#pragma unroll
  for (int p = 0; p < 4; ++p) {
    const int m  = p * 32 + (tid >> 3);
    const int kc = (tid & 7) * 4;
    As[m][kc + 0] = (_Float16)ra[p].x;
    As[m][kc + 1] = (_Float16)ra[p].y;
    As[m][kc + 2] = (_Float16)ra[p].z;
    As[m][kc + 3] = (_Float16)ra[p].w;
  }
#pragma unroll
  for (int p = 0; p < 4; ++p) {
    const int r = p * 8 + (tid >> 5);
    const int c = (tid & 31) * 4;
    WsT[c + 0][r] = (_Float16)rw[p].x;
    WsT[c + 1][r] = (_Float16)rw[p].y;
    WsT[c + 2][r] = (_Float16)rw[p].z;
    WsT[c + 3][r] = (_Float16)rw[p].w;
  }
}

template <int ACT>
__global__ __launch_bounds__(256) void gemm_wmma_kernel(
    const float* __restrict__ A, const float* __restrict__ W,
    const float* __restrict__ bias, float* __restrict__ C,
    int M, int N, int K) {
  __shared__ _Float16 As [2][BM][BK + 8];   // [buf][m][k]
  __shared__ _Float16 WsT[2][BN][BK + 8];   // [buf][n][k] (transposed W tile)

  const int tid  = threadIdx.x;
  const int lane = tid & 31;
  const int wave = tid >> 5;
  const int wm   = (wave & 3) * 32;     // wave M offset in block tile
  const int wn   = (wave >> 2) * 64;    // wave N offset in block tile
  const int rowBase = blockIdx.y * BM;
  const int colBase = blockIdx.x * BN;
  const int mrow = lane & 15;
  const int hiA  = (lane >> 4) * 8;     // A frag K swizzle + C/D M offset
  const int hiB  = (lane >> 4) * 16;    // B frag K offset

  v8f acc[2][4];
#pragma unroll
  for (int i = 0; i < 2; ++i)
#pragma unroll
    for (int j = 0; j < 4; ++j)
#pragma unroll
      for (int e = 0; e < 8; ++e) acc[i][j][e] = 0.0f;

  float4 ra[4], rw[4];
  gemm_load_tile_regs(A, W, rowBase, colBase, K, N, /*kb=*/0, tid, ra, rw);
  gemm_store_tile_lds(As[0], WsT[0], tid, ra, rw);
  __syncthreads();

  int cur = 0;
  for (int kb = 0; kb < K; kb += BK) {
    const bool has_next = (kb + BK) < K;
    // Issue next tile's global loads early; waits land at the cvt/store below.
    if (has_next)
      gemm_load_tile_regs(A, W, rowBase, colBase, K, N, kb + BK, tid, ra, rw);
    if (kb + 2 * BK < K) {
      __builtin_prefetch(A + (size_t)(rowBase + (tid >> 3)) * K + kb + 2 * BK, 0, 1);
      __builtin_prefetch(W + (size_t)(kb + 2 * BK + (tid >> 5)) * N + colBase, 0, 1);
    }

    // Fragment assembly: element s of lane half h holds k = 16h + s for both A and B.
    v16h af[2], bf[4];
#pragma unroll
    for (int mt = 0; mt < 2; ++mt) {
      const int m = wm + mt * 16 + mrow;
#pragma unroll
      for (int vg = 0; vg < 8; ++vg) {
        const int k0 = 2 * vg + 2 * hiA;
        af[mt][2 * vg]     = As[cur][m][k0];
        af[mt][2 * vg + 1] = As[cur][m][k0 + 1];
      }
    }
#pragma unroll
    for (int nt = 0; nt < 4; ++nt) {
      const int n = wn + nt * 16 + mrow;
#pragma unroll
      for (int j = 0; j < 16; ++j) bf[nt][j] = WsT[cur][n][hiB + j];
    }
#pragma unroll
    for (int mt = 0; mt < 2; ++mt)
#pragma unroll
      for (int nt = 0; nt < 4; ++nt)
        acc[mt][nt] = __builtin_amdgcn_wmma_f32_16x16x32_f16(
            false, af[mt], false, bf[nt], (short)0, acc[mt][nt], false, false);

    // Fill the alternate buffer while WMMAs drain.
    if (has_next)
      gemm_store_tile_lds(As[cur ^ 1], WsT[cur ^ 1], tid, ra, rw);
    __syncthreads();
    cur ^= 1;
  }

  // Epilogue. C/D layout: VGPR e -> M = e (lanes 0-15) or e+8 (lanes 16-31).
#pragma unroll
  for (int mt = 0; mt < 2; ++mt) {
#pragma unroll
    for (int nt = 0; nt < 4; ++nt) {
      const int n  = colBase + wn + nt * 16 + mrow;
      const float bv = bias ? bias[n] : 0.0f;
#pragma unroll
      for (int e = 0; e < 8; ++e) {
        const int m = rowBase + wm + mt * 16 + e + hiA;
        float v = acc[mt][nt][e] + bv;
        if (ACT == 1) v = v * fast_sigmoid(v);                               // silu
        else if (ACT == 2) v = 0.5f * v * (1.0f + erff(v * 0.70710678118f)); // gelu
        C[(size_t)m * N + n] = v;
      }
    }
  }
}

// ---------------------------------------------------------------------------
// Embedding gather + LayerNorm. One block per token.
// ---------------------------------------------------------------------------
__global__ __launch_bounds__(256) void embed_ln_kernel(
    const int* __restrict__ ids, const float* __restrict__ we,
    const float* __restrict__ te, const float* __restrict__ pe,
    const float* __restrict__ w, const float* __restrict__ bias,
    float* __restrict__ out) {
  __shared__ float red[256];
  const int row = blockIdx.x;
  const int tid = threadIdx.x;
  const int l   = row % L_;
  const int id  = ids[row];
  float vv[3];
  float s = 0.0f;
#pragma unroll
  for (int j = 0; j < 3; ++j) {
    const int c = tid + j * 256;
    vv[j] = we[(size_t)id * HID_ + c] + te[c] + pe[(size_t)l * HID_ + c];
    s += vv[j];
  }
  const float mean = block_reduce_sum(s, red) * (1.0f / HID_);
  float var = 0.0f;
#pragma unroll
  for (int j = 0; j < 3; ++j) { const float d = vv[j] - mean; var = fmaf(d, d, var); }
  const float rstd = rsqrtf(block_reduce_sum(var, red) * (1.0f / HID_) + 1e-12f);
#pragma unroll
  for (int j = 0; j < 3; ++j) {
    const int c = tid + j * 256;
    out[(size_t)row * HID_ + c] = (vv[j] - mean) * rstd * w[c] + bias[c];
  }
}

// ---------------------------------------------------------------------------
// Residual add + LayerNorm. One block per token.
// ---------------------------------------------------------------------------
__global__ __launch_bounds__(256) void add_ln_kernel(
    const float* __restrict__ res, const float* __restrict__ y,
    const float* __restrict__ w, const float* __restrict__ bias,
    float* __restrict__ out) {
  __shared__ float red[256];
  const int row = blockIdx.x;
  const int tid = threadIdx.x;
  float vv[3];
  float s = 0.0f;
#pragma unroll
  for (int j = 0; j < 3; ++j) {
    const int c = tid + j * 256;
    vv[j] = res[(size_t)row * HID_ + c] + y[(size_t)row * HID_ + c];
    s += vv[j];
  }
  const float mean = block_reduce_sum(s, red) * (1.0f / HID_);
  float var = 0.0f;
#pragma unroll
  for (int j = 0; j < 3; ++j) { const float d = vv[j] - mean; var = fmaf(d, d, var); }
  const float rstd = rsqrtf(block_reduce_sum(var, red) * (1.0f / HID_) + 1e-12f);
#pragma unroll
  for (int j = 0; j < 3; ++j) {
    const int c = tid + j * 256;
    out[(size_t)row * HID_ + c] = (vv[j] - mean) * rstd * w[c] + bias[c];
  }
}

// ---------------------------------------------------------------------------
// Gate projections: fd/sd (sigmoid with bias), fg/sg (sigmoid). N=12 is tiny,
// so one thread per (token, head) computing all four dots with one x pass.
// Outputs stored [b, h, l] for the delta kernel.
// ---------------------------------------------------------------------------
__global__ __launch_bounds__(256) void gates_kernel(
    const float* __restrict__ x,
    const float* __restrict__ fdw, const float* __restrict__ fdbias,
    const float* __restrict__ sdw, const float* __restrict__ sdbias,
    const float* __restrict__ fgw, const float* __restrict__ sgw,
    float* __restrict__ fd, float* __restrict__ sd,
    float* __restrict__ fg, float* __restrict__ sg) {
  const int id  = blockIdx.x * 256 + threadIdx.x;   // [0, BL_*NH_)
  const int row = id / NH_;
  const int h   = id % NH_;
  const float* xr = x + (size_t)row * HID_;
  float a0 = 0.f, a1 = 0.f, a2 = 0.f, a3 = 0.f;
  for (int k = 0; k < HID_; ++k) {
    const float xv = xr[k];
    a0 = fmaf(xv, fdw[k * NH_ + h], a0);
    a1 = fmaf(xv, sdw[k * NH_ + h], a1);
    a2 = fmaf(xv, fgw[k * NH_ + h], a2);
    a3 = fmaf(xv, sgw[k * NH_ + h], a3);
  }
  const int b = row / L_, l = row % L_;
  const size_t o = ((size_t)b * NH_ + h) * L_ + l;
  fd[o] = fast_sigmoid(a0 + fdbias[h]);
  sd[o] = fast_sigmoid(a1 + sdbias[h]);
  fg[o] = fast_sigmoid(a2);
  sg[o] = fast_sigmoid(a3);
}

// ---------------------------------------------------------------------------
// acc[0..63] = row[0..63] @ S (64x64, LDS), fp32.
// ---------------------------------------------------------------------------
__device__ __forceinline__ void row_matvec(const float* __restrict__ row,
                                           const float* __restrict__ S,
                                           float* acc) {
#pragma unroll
  for (int n = 0; n < DH_; ++n) acc[n] = 0.0f;
  for (int kk = 0; kk < DH_; ++kk) {
    const float a = row[kk];
    const float* Sr = S + kk * DH_;
#pragma unroll
    for (int n = 0; n < DH_; ++n) acc[n] = fmaf(a, Sr[n], acc[n]);
  }
}

// ---------------------------------------------------------------------------
// Delta rule recurrence. One workgroup per (b,h); 256 threads = one per chunk
// row. Sf/Ss (64x64 fp32) live in LDS; normalized state is never materialized:
// S_normalized = S * inv, inv = 1/(||S||_F + eps) carried as a scalar.
// uf/us chunk scratch lives in a per-workgroup global region (Ufg/Usg); the
// "of" partial aliases Usg (dead before us is written, same-thread rows).
// Per-head RMS norm * onorm_w is fused into the o write.
// ---------------------------------------------------------------------------
__global__ __launch_bounds__(256) void delta_rule_kernel(
    const float* __restrict__ q, const float* __restrict__ k,
    const float* __restrict__ v,
    const float* __restrict__ fd, const float* __restrict__ sd,
    const float* __restrict__ fg, const float* __restrict__ sg,
    const float* __restrict__ onorm,
    float* __restrict__ Ufg_base, float* __restrict__ Usg_base,
    float* __restrict__ o) {
  __shared__ float Sf[DH_ * DH_];
  __shared__ float Ss[DH_ * DH_];
  __shared__ float red[256];

  const int tid = threadIdx.x;
  const int bh  = blockIdx.x;
  const int b   = bh / NH_;
  const int h   = bh % NH_;
  const float* qg = q + (size_t)b * L_ * HID_ + h * DH_;
  const float* kg = k + (size_t)b * L_ * HID_ + h * DH_;
  const float* vg = v + (size_t)b * L_ * HID_ + h * DH_;
  float*       og = o + (size_t)b * L_ * HID_ + h * DH_;
  const float* fdg = fd + (size_t)bh * L_;
  const float* sdg = sd + (size_t)bh * L_;
  const float* fgg = fg + (size_t)bh * L_;
  const float* sgg = sg + (size_t)bh * L_;
  float* Ufg  = Ufg_base + (size_t)bh * CHUNK_ * DH_;
  float* Usg  = Usg_base + (size_t)bh * CHUNK_ * DH_;
  float* Obuf = Usg;  // alias: dead before Usg is written

#pragma unroll
  for (int j = 0; j < 16; ++j) { Sf[tid * 16 + j] = 0.0f; Ss[tid * 16 + j] = 0.0f; }
  __syncthreads();

  for (int c = 0; c < L_ / CHUNK_; ++c) {
    const int cs = c * CHUNK_;

    // chunk-mean decays, raised to CHUNK power
    const float fdm = block_reduce_sum(fdg[cs + tid], red) * (1.0f / CHUNK_);
    const float sdm = block_reduce_sum(sdg[cs + tid], red) * (1.0f / CHUNK_);
    const float dff = powf(fdm, (float)CHUNK_);
    const float dss = powf(sdm, (float)CHUNK_);
#pragma unroll
    for (int j = 0; j < 16; ++j) { Sf[tid * 16 + j] *= dff; Ss[tid * 16 + j] *= dss; }
    __syncthreads();

    // Frobenius norms -> scalar inverses
    float pf = 0.f, ps = 0.f;
#pragma unroll
    for (int j = 0; j < 16; ++j) {
      const float a = Sf[tid * 16 + j]; pf = fmaf(a, a, pf);
      const float e = Ss[tid * 16 + j]; ps = fmaf(e, e, ps);
    }
    const float invf = 1.0f / (sqrtf(block_reduce_sum(pf, red)) + 1e-8f);
    const float invs = 1.0f / (sqrtf(block_reduce_sum(ps, red)) + 1e-8f);

    const int r = tid;  // this thread's row in the chunk
    const float* qrow = qg + (size_t)(cs + r) * HID_;
    const float* krow = kg + (size_t)(cs + r) * HID_;
    const float* vrow = vg + (size_t)(cs + r) * HID_;
    const float fgr = fgg[cs + r];
    const float sgr = sgg[cs + r];
    float accv[DH_];

    // o = ALPHA*fg*(q@Sfr) + (1-ALPHA)*sg*(q@Ssr), then RMS-norm * onorm
    row_matvec(qrow, Sf, accv);
    {
      const float sc = ALPHA_ * fgr * invf;
#pragma unroll
      for (int n = 0; n < DH_; ++n) Obuf[r * DH_ + n] = sc * accv[n];
    }
    row_matvec(qrow, Ss, accv);
    {
      const float sc = (1.0f - ALPHA_) * sgr * invs;
      float ss = 0.f;
#pragma unroll
      for (int n = 0; n < DH_; ++n) {
        const float ov = Obuf[r * DH_ + n] + sc * accv[n];
        accv[n] = ov;
        ss = fmaf(ov, ov, ss);
      }
      const float rinv = rsqrtf(ss * (1.0f / DH_) + 1e-6f);
#pragma unroll
      for (int n = 0; n < DH_; ++n)
        og[(size_t)(cs + r) * HID_ + n] = accv[n] * rinv * onorm[n];
    }

    // uf = v - (k@Sf)*invf ; us = v - (k@Ss)*invs
    row_matvec(krow, Sf, accv);
#pragma unroll
    for (int n = 0; n < DH_; ++n) Ufg[r * DH_ + n] = vrow[n] - invf * accv[n];
    row_matvec(krow, Ss, accv);
#pragma unroll
    for (int n = 0; n < DH_; ++n) Usg[r * DH_ + n] = vrow[n] - invs * accv[n];

    __threadfence();   // make Ufg/Usg visible across the workgroup
    __syncthreads();

    // S += k^T @ u : thread (kk, nq) owns 16 columns of state row kk
    const int kk = tid >> 2;
    const int nq = (tid & 3) * 16;
    float accf[16], accs[16];
#pragma unroll
    for (int j = 0; j < 16; ++j) { accf[j] = 0.f; accs[j] = 0.f; }
    for (int rr = 0; rr < CHUNK_; ++rr) {
      const float kv = kg[(size_t)(cs + rr) * HID_ + kk];
      const float* ur = Ufg + rr * DH_ + nq;
      const float* sr = Usg + rr * DH_ + nq;
#pragma unroll
      for (int j = 0; j < 16; ++j) {
        accf[j] = fmaf(kv, ur[j], accf[j]);
        accs[j] = fmaf(kv, sr[j], accs[j]);
      }
    }
#pragma unroll
    for (int j = 0; j < 16; ++j) {
      Sf[kk * DH_ + nq + j] += accf[j];
      Ss[kk * DH_ + nq + j] += accs[j];
    }
    __syncthreads();

    // normalize stored state
    pf = 0.f; ps = 0.f;
#pragma unroll
    for (int j = 0; j < 16; ++j) {
      const float a = Sf[tid * 16 + j]; pf = fmaf(a, a, pf);
      const float e = Ss[tid * 16 + j]; ps = fmaf(e, e, ps);
    }
    const float scf = 1.0f / (sqrtf(block_reduce_sum(pf, red)) + 1e-8f);
    const float scs = 1.0f / (sqrtf(block_reduce_sum(ps, red)) + 1e-8f);
#pragma unroll
    for (int j = 0; j < 16; ++j) { Sf[tid * 16 + j] *= scf; Ss[tid * 16 + j] *= scs; }
    __syncthreads();
  }
}

// ---------------------------------------------------------------------------
// Host-side orchestration. Workspace needed: ~194 MiB of fp32.
// ---------------------------------------------------------------------------
extern "C" void kernel_launch(void* const* d_in, const int* in_sizes, int n_in,
                              void* d_out, int out_size, void* d_ws, size_t ws_size,
                              hipStream_t stream) {
  (void)in_sizes; (void)n_in; (void)out_size; (void)ws_size;
  const int*   input_ids = (const int*)  d_in[0];
  const float* word_emb  = (const float*)d_in[1];
  const float* tok_emb   = (const float*)d_in[2];
  const float* pos_emb   = (const float*)d_in[3];
  const float* emb_ln_w  = (const float*)d_in[4];
  const float* emb_ln_b  = (const float*)d_in[5];
  const float* q_w       = (const float*)d_in[6];
  const float* k_w       = (const float*)d_in[7];
  const float* v_w       = (const float*)d_in[8];
  const float* fd_w      = (const float*)d_in[9];
  const float* fd_b      = (const float*)d_in[10];
  const float* sd_w      = (const float*)d_in[11];
  const float* sd_b      = (const float*)d_in[12];
  const float* fg_w      = (const float*)d_in[13];
  const float* sg_w      = (const float*)d_in[14];
  const float* onorm_w   = (const float*)d_in[15];
  const float* o_w       = (const float*)d_in[16];
  const float* attn_ln_w = (const float*)d_in[17];
  const float* attn_ln_b = (const float*)d_in[18];
  const float* ffn_w1    = (const float*)d_in[19];
  const float* ffn_b1    = (const float*)d_in[20];
  const float* ffn_w2    = (const float*)d_in[21];
  const float* ffn_b2    = (const float*)d_in[22];
  const float* ffn_ln_w  = (const float*)d_in[23];
  const float* ffn_ln_b  = (const float*)d_in[24];

  float* ws = (float*)d_ws;
  size_t off = 0;
  const size_t XSZ = (size_t)BL_ * HID_;
  float* x0  = ws + off; off += XSZ;
  float* x1  = ws + off; off += XSZ;
  float* qb  = ws + off; off += XSZ;
  float* kb  = ws + off; off += XSZ;
  float* vb  = ws + off; off += XSZ;
  float* ob  = ws + off; off += XSZ;
  float* hff = qb;  // FFN hidden [BL_,3072] aliases qb..ob (exactly 4*XSZ), dead by then
  float* t0  = ws + off; off += XSZ;
  const size_t GSZ = (size_t)B_ * NH_ * L_;
  float* fdb = ws + off; off += GSZ;
  float* sdb = ws + off; off += GSZ;
  float* fgb = ws + off; off += GSZ;
  float* sgb = ws + off; off += GSZ;
  const size_t USZ = (size_t)B_ * NH_ * CHUNK_ * DH_;
  float* Ufg = ws + off; off += USZ;
  float* Usg = ws + off; off += USZ;

  const dim3 blk(256);
  const dim3 gHID(HID_ / BN, BL_ / BM);   // 6 x 64
  const dim3 gFFN(FFN_ / BN, BL_ / BM);   // 24 x 64

  embed_ln_kernel<<<BL_, blk, 0, stream>>>(input_ids, word_emb, tok_emb, pos_emb,
                                           emb_ln_w, emb_ln_b, x0);

  for (int i = 0; i < NL_; ++i) {
    const float* qwi = q_w + (size_t)i * HID_ * HID_;
    const float* kwi = k_w + (size_t)i * HID_ * HID_;
    const float* vwi = v_w + (size_t)i * HID_ * HID_;
    const float* owi = o_w + (size_t)i * HID_ * HID_;

    gemm_wmma_kernel<1><<<gHID, blk, 0, stream>>>(x0, qwi, nullptr, qb, BL_, HID_, HID_);
    gemm_wmma_kernel<1><<<gHID, blk, 0, stream>>>(x0, kwi, nullptr, kb, BL_, HID_, HID_);
    gemm_wmma_kernel<1><<<gHID, blk, 0, stream>>>(x0, vwi, nullptr, vb, BL_, HID_, HID_);

    gates_kernel<<<(BL_ * NH_) / 256, blk, 0, stream>>>(
        x0, fd_w + (size_t)i * HID_ * NH_, fd_b + (size_t)i * NH_,
        sd_w + (size_t)i * HID_ * NH_, sd_b + (size_t)i * NH_,
        fg_w + (size_t)i * HID_ * NH_, sg_w + (size_t)i * HID_ * NH_,
        fdb, sdb, fgb, sgb);

    delta_rule_kernel<<<B_ * NH_, blk, 0, stream>>>(
        qb, kb, vb, fdb, sdb, fgb, sgb, onorm_w + (size_t)i * DH_, Ufg, Usg, ob);

    gemm_wmma_kernel<0><<<gHID, blk, 0, stream>>>(ob, owi, nullptr, t0, BL_, HID_, HID_);
    add_ln_kernel<<<BL_, blk, 0, stream>>>(x0, t0, attn_ln_w + (size_t)i * HID_,
                                           attn_ln_b + (size_t)i * HID_, x1);

    gemm_wmma_kernel<2><<<gFFN, blk, 0, stream>>>(
        x1, ffn_w1 + (size_t)i * HID_ * FFN_, ffn_b1 + (size_t)i * FFN_,
        hff, BL_, FFN_, HID_);
    gemm_wmma_kernel<0><<<gHID, blk, 0, stream>>>(
        hff, ffn_w2 + (size_t)i * FFN_ * HID_, ffn_b2 + (size_t)i * HID_,
        t0, BL_, HID_, FFN_);

    float* xout = (i == NL_ - 1) ? (float*)d_out : x0;
    add_ln_kernel<<<BL_, blk, 0, stream>>>(x1, t0, ffn_ln_w + (size_t)i * HID_,
                                           ffn_ln_b + (size_t)i * HID_, xout);
  }
}
